// MultiHeadAttention_33303176413857
// MI455X (gfx1250) — matrix operand order repaired and checked by reading, not compiled, not run on hardware
//
#include <hip/hip_runtime.h>
#include <hip/hip_bf16.h>

// ---------------------------------------------------------------------------
// MultiHeadAttention forward (B=4, S=2048, D=1024, H=16, dk=dv=64) for gfx1250
// bf16 WMMA (v_wmma_f32_16x16x32_bf16) for all GEMMs, fp32 softmax/LayerNorm.
// Async global->LDS staging (ASYNCcnt) where tiles are raw bf16 copies.
// ---------------------------------------------------------------------------

#define B_SZ   4
#define SEQ    2048
#define DMODEL 1024
#define NHEAD  16
#define DHEAD  64

#define KC   32          // K-chunk per WMMA step
#define LDSP 40          // LDS row pitch in bf16: 80B rows -> 16B aligned for B128
                         // async writes; 20-bank stride -> conflict-free frag reads

typedef __attribute__((ext_vector_type(16))) __bf16 v16bf;
typedef __attribute__((ext_vector_type(8)))  float  v8f;
typedef __attribute__((ext_vector_type(4)))  int    v4i;

union FragAB { v16bf v; unsigned u[8]; };

// -------- CDNA5 async global->LDS path (guarded; fallback = VALU copy) ------
#if defined(__has_builtin)
#if __has_builtin(__builtin_amdgcn_global_load_async_to_lds_b128) && \
    __has_builtin(__builtin_amdgcn_s_wait_asynccnt)
#define HAVE_ASYNC_LDS 1
#endif
#endif
#ifndef HAVE_ASYNC_LDS
#define HAVE_ASYNC_LDS 0
#endif

#if HAVE_ASYNC_LDS
typedef __attribute__((address_space(1))) v4i gv4i;   // global int4
typedef __attribute__((address_space(3))) v4i lv4i;   // LDS int4
#endif

// ---------------- LDS staging helpers (256 threads) ----------------

// dst[r][c] = (bf16) src[(row0+r)*ld + k0 + c], r in [0,ROWS), c in [0,KC)
template<int ROWS, typename T>
__device__ inline void lds_fill_rowmajor(__bf16* dst, const T* __restrict__ src,
                                         int ld, int row0, int k0) {
    constexpr int TPR = 256 / ROWS;   // threads per row
    constexpr int CPT = KC / TPR;     // cols per thread
    const int t = threadIdx.x;
    const int r = t / TPR;
    const int c = (t % TPR) * CPT;
    const T* s = src + (size_t)(row0 + r) * ld + k0 + c;
    __bf16* d = dst + r * LDSP + c;
#pragma unroll
    for (int i = 0; i < CPT; ++i) d[i] = (__bf16)s[i];
}

// dst[n][k] = (bf16) src[(k0+k)*ld + n0 + n]  (builds B^T tile from row-major B)
template<int BN, typename T>
__device__ inline void lds_fill_transpose(__bf16* dst, const T* __restrict__ src,
                                          int ld, int k0, int n0) {
    constexpr int TPK = 256 / KC;     // 8 threads per k-row
    constexpr int NPT = BN / TPK;     // n's per thread
    const int t = threadIdx.x;
    const int k = t / TPK;
    const int n = (t % TPK) * NPT;
    const T* s = src + (size_t)(k0 + k) * ld + n0 + n;
#pragma unroll
    for (int i = 0; i < NPT; ++i) dst[(n + i) * LDSP + k] = (__bf16)s[i];
}

// Raw bf16 ROWS x KC tile copy, 16 bytes per issue, no VGPR round trip.
template<int ROWS>
__device__ inline void lds_fill_bf16_direct(__bf16* dst, const __bf16* __restrict__ src,
                                            int ld, int row0, int k0) {
#if HAVE_ASYNC_LDS
    constexpr int PER_T = (ROWS * 4) / 256;   // 4 x 16B chunks per 64B row
    const int t = threadIdx.x;
#pragma unroll
    for (int i = 0; i < PER_T; ++i) {
        const int idx = t + i * 256;
        const int r = idx >> 2, c = (idx & 3) * 8;
        const __bf16* g = src + (size_t)(row0 + r) * ld + k0 + c;
        __bf16* l = dst + r * LDSP + c;
        __builtin_amdgcn_global_load_async_to_lds_b128((gv4i*)g, (lv4i*)l, 0, 0);
    }
#else
    lds_fill_rowmajor<ROWS, __bf16>(dst, src, ld, row0, k0);
#endif
}

__device__ inline void async_tile_wait() {
#if HAVE_ASYNC_LDS
    __builtin_amdgcn_s_wait_asynccnt(0);
#endif
}

// ---------------- WMMA fragment loads (CDNA5 wave32 layouts) ----------------

// A 16x32 bf16: lanes 0-15 row M=lane hold K=0..7 (VGPR0-3), K=16..23 (VGPR4-7);
// lanes 16-31 same rows hold K=8..15 and K=24..31.
__device__ inline v16bf load_frag_a(const __bf16* lds, int rowBase) {
    const int lane = threadIdx.x & 31;
    const int r  = lane & 15;
    const int hi = lane >> 4;
    const __bf16* base = lds + (rowBase + r) * LDSP;
    FragAB f;
#pragma unroll
    for (int j = 0; j < 8; ++j) {
        const int k = ((j >= 4) ? 16 : 0) + hi * 8 + (j & 3) * 2;  // K, K+1 packed
        f.u[j] = *(const unsigned*)(base + k);
    }
    return f.v;
}

// B 32x16 bf16 from B^T tile: lane = column N; lanes 0-15 hold K=0..7, K=16..23,
// lanes 16-31 hold K=8..15, K=24..31 (same K order as A).
__device__ inline v16bf load_frag_b(const __bf16* ldsBt, int colBase) {
    const int lane = threadIdx.x & 31;
    const int n    = lane & 15;
    const int kOff = (lane >> 4) * 8;
    const __bf16* base = ldsBt + (colBase + n) * LDSP;
    FragAB f;
#pragma unroll
    for (int j = 0; j < 8; ++j) f.u[j] = *(const unsigned*)(base + kOff + 2 * j + ((j >= 4) ? 8 : 0));
    return f.v;
}

#define WMMA_BF16(a, b, c) \
    __builtin_amdgcn_wmma_f32_16x16x32_bf16(false, (a), false, (b), (short)0, (c), false, false)

// ---------------- Kernel 1: fused Q/K/V projections ----------------
// grid = (8192/128, 1024/128, 3); X[8192x1024] @ W[1024x1024] -> [B,H,S,64] bf16
// Block tile 128x128, 4x2 waves, 32x64 per wave (8 WMMA / K-chunk).
__global__ __launch_bounds__(256) void k_proj(
    const float* __restrict__ q, const float* __restrict__ k, const float* __restrict__ v,
    const float* __restrict__ wq, const float* __restrict__ wk, const float* __restrict__ wv,
    __bf16* __restrict__ qh, __bf16* __restrict__ kh, __bf16* __restrict__ vh) {
    __shared__ __align__(16) __bf16 ldsA[128 * LDSP];
    __shared__ __align__(16) __bf16 ldsB[128 * LDSP];

    const float* X; const float* W; __bf16* dst;
    if (blockIdx.z == 0)      { X = q; W = wq; dst = qh; }
    else if (blockIdx.z == 1) { X = k; W = wk; dst = kh; }
    else                      { X = v; W = wv; dst = vh; }

    const int row0 = blockIdx.x * 128;
    const int n0   = blockIdx.y * 128;
    const int wave = threadIdx.x >> 5, lane = threadIdx.x & 31;
    const int wm = wave >> 1, wn = wave & 1;    // 4 x 2 wave grid

    v8f acc[2][4] = {};
    for (int k0 = 0; k0 < DMODEL; k0 += KC) {
        if (k0 + KC < DMODEL) {   // prefetch next weight chunk (global_prefetch_b8)
            __builtin_prefetch(W + (size_t)(k0 + KC + (threadIdx.x >> 3)) * DMODEL
                                 + n0 + (threadIdx.x & 7) * 16, 0, 1);
        }
        lds_fill_rowmajor<128, float>(ldsA, X, DMODEL, row0, k0);
        lds_fill_transpose<128, float>(ldsB, W, DMODEL, k0, n0);
        __syncthreads();
        v16bf a0 = load_frag_a(ldsA, wm * 32);
        v16bf a1 = load_frag_a(ldsA, wm * 32 + 16);
#pragma unroll
        for (int t = 0; t < 4; ++t) {
            v16bf b = load_frag_b(ldsB, wn * 64 + t * 16);
            acc[0][t] = WMMA_BF16(a0, b, acc[0][t]);
            acc[1][t] = WMMA_BF16(a1, b, acc[1][t]);
        }
        __syncthreads();
    }

    const int rOff = (lane >> 4) * 8;
    const int nl   = lane & 15;
#pragma unroll
    for (int mh = 0; mh < 2; ++mh) {
#pragma unroll
        for (int t = 0; t < 4; ++t) {
            const int gcol = n0 + wn * 64 + t * 16 + nl;
            const int h = gcol >> 6, d = gcol & 63;
#pragma unroll
            for (int r = 0; r < 8; ++r) {
                const int grow = row0 + wm * 32 + mh * 16 + rOff + r;
                const int b = grow >> 11, s = grow & (SEQ - 1);
                dst[(((size_t)(b * NHEAD + h) * SEQ + s) << 6) + d] = (__bf16)acc[mh][t][r];
            }
        }
    }
}

// ---------------- Kernel 2: scores = qh @ kh^T * 1/sqrt(dk) ----------------
// grid = (2048/128, 2048/128, B*H). Both tiles are raw bf16 row-major copies:
// fully async-staged (kh rows [key][d] already are the B^T layout).
__global__ __launch_bounds__(256) void k_scores(
    const __bf16* __restrict__ qh, const __bf16* __restrict__ kh, float* __restrict__ attn) {
    __shared__ __align__(16) __bf16 ldsA[128 * LDSP];
    __shared__ __align__(16) __bf16 ldsB[128 * LDSP];

    const int z = blockIdx.z;
    const __bf16* Q  = qh + (size_t)z * SEQ * DHEAD;
    const __bf16* Kp = kh + (size_t)z * SEQ * DHEAD;
    float* A = attn + (size_t)z * SEQ * SEQ;

    const int row0 = blockIdx.x * 128;
    const int n0   = blockIdx.y * 128;
    const int wave = threadIdx.x >> 5, lane = threadIdx.x & 31;
    const int wm = wave >> 1, wn = wave & 1;

    v8f acc[2][4] = {};
    for (int k0 = 0; k0 < DHEAD; k0 += KC) {
        lds_fill_bf16_direct<128>(ldsA, Q, DHEAD, row0, k0);
        lds_fill_bf16_direct<128>(ldsB, Kp, DHEAD, n0, k0);
        async_tile_wait();
        __syncthreads();
        v16bf a0 = load_frag_a(ldsA, wm * 32);
        v16bf a1 = load_frag_a(ldsA, wm * 32 + 16);
#pragma unroll
        for (int t = 0; t < 4; ++t) {
            v16bf b = load_frag_b(ldsB, wn * 64 + t * 16);
            acc[0][t] = WMMA_BF16(a0, b, acc[0][t]);
            acc[1][t] = WMMA_BF16(a1, b, acc[1][t]);
        }
        __syncthreads();
    }

    const int rOff = (lane >> 4) * 8;
    const int nl   = lane & 15;
#pragma unroll
    for (int mh = 0; mh < 2; ++mh) {
#pragma unroll
        for (int t = 0; t < 4; ++t) {
            const int gcol = n0 + wn * 64 + t * 16 + nl;
#pragma unroll
            for (int r = 0; r < 8; ++r) {
                const int grow = row0 + wm * 32 + mh * 16 + rOff + r;
                A[(size_t)grow * SEQ + gcol] = acc[mh][t][r] * 0.125f;  // 1/sqrt(64)
            }
        }
    }
}

// ---------------- Kernel 3: row softmax over 2048, in place ----------------
__global__ __launch_bounds__(256) void k_softmax(float* __restrict__ attn) {
    __shared__ float red[8];
    float* p = attn + (size_t)blockIdx.x * SEQ;
    const int t = threadIdx.x;

    float x[8];
#pragma unroll
    for (int i = 0; i < 8; ++i) x[i] = p[t + i * 256];

    float m = x[0];
#pragma unroll
    for (int i = 1; i < 8; ++i) m = fmaxf(m, x[i]);
#pragma unroll
    for (int off = 16; off; off >>= 1) m = fmaxf(m, __shfl_xor(m, off, 32));
    if ((t & 31) == 0) red[t >> 5] = m;
    __syncthreads();
    m = red[0];
#pragma unroll
    for (int j = 1; j < 8; ++j) m = fmaxf(m, red[j]);
    __syncthreads();

    float s = 0.f;
#pragma unroll
    for (int i = 0; i < 8; ++i) { x[i] = __expf(x[i] - m); s += x[i]; }
#pragma unroll
    for (int off = 16; off; off >>= 1) s += __shfl_xor(s, off, 32);
    if ((t & 31) == 0) red[t >> 5] = s;
    __syncthreads();
    s = red[0];
#pragma unroll
    for (int j = 1; j < 8; ++j) s += red[j];

    const float inv = 1.0f / s;
#pragma unroll
    for (int i = 0; i < 8; ++i) p[t + i * 256] = x[i] * inv;
}

// ---------------- Kernel 4: ctx = attn @ vh  -> [B,S,H*64] bf16 ----------------
// grid = (2048/128, 1, B*H); BM=128, BN=64 (8x1 waves); K = 2048
__global__ __launch_bounds__(256) void k_ctx(
    const float* __restrict__ attn, const __bf16* __restrict__ vh, __bf16* __restrict__ ctx) {
    __shared__ __align__(16) __bf16 ldsA[128 * LDSP];
    __shared__ __align__(16) __bf16 ldsB[64 * LDSP];

    const int z = blockIdx.z;
    const float* A  = attn + (size_t)z * SEQ * SEQ;
    const __bf16* V = vh + (size_t)z * SEQ * DHEAD;

    const int row0 = blockIdx.x * 128;
    const int wave = threadIdx.x >> 5, lane = threadIdx.x & 31;
    const int wm = wave;                         // 8 x 1 wave grid

    v8f acc[4] = {};
    for (int k0 = 0; k0 < SEQ; k0 += KC) {
        lds_fill_rowmajor<128, float>(ldsA, A, SEQ, row0, k0);   // fp32 -> bf16
        lds_fill_transpose<64, __bf16>(ldsB, V, DHEAD, k0, 0);   // V^T tile
        __syncthreads();
        v16bf a = load_frag_a(ldsA, wm * 16);
#pragma unroll
        for (int t = 0; t < 4; ++t) {
            v16bf b = load_frag_b(ldsB, t * 16);
            acc[t] = WMMA_BF16(a, b, acc[t]);
        }
        __syncthreads();
    }

    const int rOff = (lane >> 4) * 8;
    const int nl   = lane & 15;
    const int b = z >> 4, h = z & 15;
#pragma unroll
    for (int t = 0; t < 4; ++t) {
        const int d = t * 16 + nl;
#pragma unroll
        for (int r = 0; r < 8; ++r) {
            const int s = row0 + wm * 16 + rOff + r;
            ctx[((size_t)(b * SEQ + s)) * DMODEL + h * DHEAD + d] = (__bf16)acc[t][r];
        }
    }
}

// ---------------- Kernel 5: out = ctx @ w_o + residual(q) ----------------
// grid = (8192/128, 1024/128); A tile async-staged (ctx is bf16 row-major).
__global__ __launch_bounds__(256) void k_outproj(
    const __bf16* __restrict__ ctx, const float* __restrict__ wo,
    const float* __restrict__ resid, float* __restrict__ out) {
    __shared__ __align__(16) __bf16 ldsA[128 * LDSP];
    __shared__ __align__(16) __bf16 ldsB[128 * LDSP];

    const int row0 = blockIdx.x * 128;
    const int n0   = blockIdx.y * 128;
    const int wave = threadIdx.x >> 5, lane = threadIdx.x & 31;
    const int wm = wave >> 1, wn = wave & 1;

    v8f acc[2][4] = {};
    for (int k0 = 0; k0 < DMODEL; k0 += KC) {
        if (k0 + KC < DMODEL) {
            __builtin_prefetch(wo + (size_t)(k0 + KC + (threadIdx.x >> 3)) * DMODEL
                                  + n0 + (threadIdx.x & 7) * 16, 0, 1);
        }
        lds_fill_bf16_direct<128>(ldsA, ctx, DMODEL, row0, k0);
        lds_fill_transpose<128, float>(ldsB, wo, DMODEL, k0, n0);
        async_tile_wait();
        __syncthreads();
        v16bf a0 = load_frag_a(ldsA, wm * 32);
        v16bf a1 = load_frag_a(ldsA, wm * 32 + 16);
#pragma unroll
        for (int t = 0; t < 4; ++t) {
            v16bf b = load_frag_b(ldsB, wn * 64 + t * 16);
            acc[0][t] = WMMA_BF16(a0, b, acc[0][t]);
            acc[1][t] = WMMA_BF16(a1, b, acc[1][t]);
        }
        __syncthreads();
    }

    const int rOff = (lane >> 4) * 8;
    const int nl   = lane & 15;
#pragma unroll
    for (int mh = 0; mh < 2; ++mh) {
#pragma unroll
        for (int t = 0; t < 4; ++t) {
            const int gcol = n0 + wn * 64 + t * 16 + nl;
#pragma unroll
            for (int r = 0; r < 8; ++r) {
                const int grow = row0 + wm * 32 + mh * 16 + rOff + r;
                const size_t idx = (size_t)grow * DMODEL + gcol;
                out[idx] = acc[mh][t][r] + resid[idx];
            }
        }
    }
}

// ---------------- Kernel 6: LayerNorm over 1024, in place ----------------
__global__ __launch_bounds__(256) void k_ln(
    float* __restrict__ out, const float* __restrict__ gamma, const float* __restrict__ beta) {
    __shared__ float red[8];
    float* p = out + (size_t)blockIdx.x * DMODEL;
    const int t = threadIdx.x;

    float x[4];
#pragma unroll
    for (int i = 0; i < 4; ++i) x[i] = p[t + i * 256];

    float s = x[0] + x[1] + x[2] + x[3];
#pragma unroll
    for (int off = 16; off; off >>= 1) s += __shfl_xor(s, off, 32);
    if ((t & 31) == 0) red[t >> 5] = s;
    __syncthreads();
    s = red[0];
#pragma unroll
    for (int j = 1; j < 8; ++j) s += red[j];
    const float mu = s * (1.0f / DMODEL);
    __syncthreads();

    float q2 = 0.f;
#pragma unroll
    for (int i = 0; i < 4; ++i) { const float d = x[i] - mu; q2 += d * d; }
#pragma unroll
    for (int off = 16; off; off >>= 1) q2 += __shfl_xor(q2, off, 32);
    if ((t & 31) == 0) red[t >> 5] = q2;
    __syncthreads();
    q2 = red[0];
#pragma unroll
    for (int j = 1; j < 8; ++j) q2 += red[j];
    const float rstd = rsqrtf(q2 * (1.0f / DMODEL) + 1e-6f);

#pragma unroll
    for (int i = 0; i < 4; ++i) {
        const int c = t + i * 256;
        p[c] = (x[i] - mu) * rstd * gamma[c] + beta[c];
    }
}

// ---------------- launch ----------------
extern "C" void kernel_launch(void* const* d_in, const int* in_sizes, int n_in,
                              void* d_out, int out_size, void* d_ws, size_t ws_size,
                              hipStream_t stream) {
    const float* q  = (const float*)d_in[0];
    const float* k  = (const float*)d_in[1];
    const float* v  = (const float*)d_in[2];
    const float* wq = (const float*)d_in[3];
    const float* wk = (const float*)d_in[4];
    const float* wv = (const float*)d_in[5];
    const float* wo = (const float*)d_in[6];
    const float* ga = (const float*)d_in[7];
    const float* be = (const float*)d_in[8];

    float* out  = (float*)d_out;                       // [B,S,1024]
    float* attn = out + (size_t)B_SZ * SEQ * DMODEL;   // [B,H,S,S]

    const size_t nBSD = (size_t)B_SZ * SEQ * DMODEL;   // 8388608
    __bf16* ws  = (__bf16*)d_ws;
    __bf16* qh  = ws;
    __bf16* kh  = ws + nBSD;
    __bf16* vh  = ws + 2 * nBSD;
    __bf16* ctx = ws + 3 * nBSD;

    dim3 blk(256);
    k_proj   <<<dim3(B_SZ * SEQ / 128, DMODEL / 128, 3), blk, 0, stream>>>(q, k, v, wq, wk, wv, qh, kh, vh);
    k_scores <<<dim3(SEQ / 128, SEQ / 128, B_SZ * NHEAD), blk, 0, stream>>>(qh, kh, attn);
    k_softmax<<<dim3(B_SZ * NHEAD * SEQ), blk, 0, stream>>>(attn);
    k_ctx    <<<dim3(SEQ / 128, 1, B_SZ * NHEAD), blk, 0, stream>>>(attn, vh, ctx);
    k_outproj<<<dim3(B_SZ * SEQ / 128, DMODEL / 128), blk, 0, stream>>>(ctx, wo, q, out);
    k_ln     <<<dim3(B_SZ * SEQ), blk, 0, stream>>>(out, ga, be);
}
